// FireLSTM_56307021250793
// MI455X (gfx1250) — matrix operand order repaired and checked by reading, not compiled, not run on hardware
//
#include <hip/hip_runtime.h>

typedef __attribute__((ext_vector_type(16))) _Float16 v16h;
typedef __attribute__((ext_vector_type(8)))  _Float16 v8h;
typedef __attribute__((ext_vector_type(4)))  _Float16 v4h;
typedef __attribute__((ext_vector_type(8)))  float    v8f;
typedef __attribute__((ext_vector_type(4)))  float    v4f;

#define TSTEPS 512
#define ROWH   104   // A-stage row length in halfs (208B: 16B aligned, bank-friendly stride)
#define NFRAG  48    // 16 N-tiles * 3 K-chunks
#define WPB    4     // 4 waves per block: each wave owns one hidden sub-tile s (16 of 64 h units)

// hardware tanh (TRANS op on gfx1250)
__device__ __forceinline__ float ftanh(float x) {
#if __has_builtin(__builtin_amdgcn_tanhf)
  return __builtin_amdgcn_tanhf(x);
#elif __has_builtin(__builtin_amdgcn_tanh_f32)
  return __builtin_amdgcn_tanh_f32(x);
#else
  return 1.0f - 2.0f * __builtin_amdgcn_rcpf(1.0f + __builtin_amdgcn_exp2f(2.88539008f * x));
#endif
}
// sigmoid with the 0.5 input scale pre-folded into the weights: sig(z) = 0.5 + 0.5*tanh(z/2)
__device__ __forceinline__ float fsig_pre(float half_z) {
  return __builtin_fmaf(0.5f, ftanh(half_z), 0.5f);
}

__device__ __forceinline__ v8f wmma_f16(v16h a, v16h b, v8f c) {
  return __builtin_amdgcn_wmma_f32_16x16x32_f16(false, a, false, b, (short)0, c, false, false);
}

union AFrag { v16h v; v8h h[2]; };
union HFrag { v16h v; _Float16 e[16]; };

__global__ __launch_bounds__(WPB * 32)
void firelstm_kernel(const float* __restrict__ x,
                     const float* __restrict__ Wih,
                     const float* __restrict__ Whh,
                     const float* __restrict__ bih,
                     const float* __restrict__ bhh,
                     const float* __restrict__ Wfc,
                     const float* __restrict__ bfc,
                     float* __restrict__ out)
{
  // B fragments in WMMA B layout. K: 0..3 = x, 4..67 = h, 68 = bias (A col 68 == 1), 69.. = 0.
  // Rows feeding the sigmoid gates (i,f,o) are pre-scaled by 0.5.
  __shared__ _Float16 wfrag[NFRAG][32][16];
  // double-buffered A stage: step t reads abuf[t&1], writes h_{t+1}/x_{t+1} into abuf[(t+1)&1]
  __shared__ _Float16 abuf[2][16][ROWH];

  const int tid  = threadIdx.x;
  const int w    = tid >> 5;     // wave id == hidden sub-tile s (0..3)
  const int lane = tid & 31;
  const int half = lane >> 4;
  const int col  = lane & 15;
  const int b0   = blockIdx.x * 16;

  // ---- stage [0.5*W|W].T + bias as f16 B fragments (gate g = ntile>>2; g==2 is 'g' gate) ----
  for (int f = w; f < NFRAG; f += WPB) {
    const int ntile = f / 3;
    const int kc    = f - ntile * 3;
    const int n     = ntile * 16 + col;           // gate column 0..255
    const float sc  = ((ntile >> 2) == 2) ? 1.0f : 0.5f;   // tanh gate unscaled
    HFrag tmp;
#pragma unroll
    for (int e = 0; e < 16; ++e) {
      const int k = kc * 32 + half * 16 + e;      // B row (K index)
      float vv = 0.0f;
      if (k < 4)        vv = Wih[n * 4 + k];
      else if (k < 68)  vv = Whh[n * 64 + (k - 4)];
      else if (k == 68) vv = bih[n] + bhh[n];     // bias row (A supplies constant 1)
      tmp.e[e] = (_Float16)(vv * sc);
    }
    *(v16h*)&wfrag[f][lane][0] = tmp.v;
  }

  // ---- init both A buffers: zeros, ones-column at K=68 ----
  if (tid < 32) {
    const int buf = tid >> 4, r = tid & 15;
#pragma unroll 4
    for (int cc = 0; cc < ROWH; ++cc) abuf[buf][r][cc] = (_Float16)0.0f;
    abuf[buf][r][68] = (_Float16)1.0f;
  }

  // ---- stage x_0 into abuf[0]; prefetch x_1 (wave 0 owns x streaming) ----
  const float* xrow = x + (size_t)(b0 + col) * (TSTEPS * 4);
  v4f xv = {};
  if (w == 0) {
    const v4f x0 = *(const v4f*)xrow;
    if (lane < 16) {
      v4h xh;
      xh[0] = (_Float16)x0[0]; xh[1] = (_Float16)x0[1];
      xh[2] = (_Float16)x0[2]; xh[3] = (_Float16)x0[3];
      *(v4h*)&abuf[0][lane][0] = xh;
    }
    xv = *(const v4f*)(xrow + 4);                 // x_1
  }
  __syncthreads();

  // ---- my 12 B fragments -> registers (96 VGPRs, loop-invariant) ----
  v16h bfr[4][3];                                 // [gate i,f,g,o][kc], ntile = gate*4 + w
#pragma unroll
  for (int g = 0; g < 4; ++g)
#pragma unroll
    for (int kc = 0; kc < 3; ++kc)
      bfr[g][kc] = *(const v16h*)&wfrag[(g * 4 + w) * 3 + kc][lane][0];

  // cell state fragment: cst[v] <-> (row m = v+8*half, hidden j = 16*w+col)
  float cst[8];
#pragma unroll
  for (int v = 0; v < 8; ++v) cst[v] = 0.0f;

  const v8f zf = {};                              // zero C -> inline 0 in v_wmma

  for (int t = 0; t < TSTEPS; ++t) {
    // A fragments for this step (full K: every wave needs all of h and x)
    AFrag a[3];
    const _Float16* arow = &abuf[t & 1][col][0];
#pragma unroll
    for (int kc = 0; kc < 3; ++kc) {
      a[kc].h[0] = *(const v8h*)(arow + half * 16 + kc * 32);
      a[kc].h[1] = *(const v8h*)(arow + half * 16 + kc * 32 + 8);
    }

    // wave 0: stage x_{t+1} into next buffer, prefetch x_{t+2}
    if (w == 0) {
      if (lane < 16 && t + 1 < TSTEPS) {
        v4h xh;
        xh[0] = (_Float16)xv[0]; xh[1] = (_Float16)xv[1];
        xh[2] = (_Float16)xv[2]; xh[3] = (_Float16)xv[3];
        *(v4h*)&abuf[(t + 1) & 1][lane][0] = xh;
      }
      if (t + 2 < TSTEPS) xv = *(const v4f*)(xrow + (size_t)(t + 2) * 4);
    }

    // 4 independent 3-deep WMMA chains (i, f, g, o) for my sub-tile
    v8f ai, af_, ag, ao;
    ai  = wmma_f16(a[0].v, bfr[0][0], zf);
    af_ = wmma_f16(a[0].v, bfr[1][0], zf);
    ag  = wmma_f16(a[0].v, bfr[2][0], zf);
    ao  = wmma_f16(a[0].v, bfr[3][0], zf);
#pragma unroll
    for (int kc = 1; kc < 3; ++kc) {
      ai  = wmma_f16(a[kc].v, bfr[0][kc], ai);
      af_ = wmma_f16(a[kc].v, bfr[1][kc], af_);
      ag  = wmma_f16(a[kc].v, bfr[2][kc], ag);
      ao  = wmma_f16(a[kc].v, bfr[3][kc], ao);
    }

    // elementwise LSTM cell update; write new h (f16) into the next A buffer
#pragma unroll
    for (int v = 0; v < 8; ++v) {
      const float iv = fsig_pre(ai[v]);           // weights pre-scaled by 0.5
      const float fv = fsig_pre(af_[v]);
      const float gv = ftanh(ag[v]);
      const float ov = fsig_pre(ao[v]);
      const float cn = __builtin_fmaf(fv, cst[v], iv * gv);
      cst[v] = cn;
      abuf[(t + 1) & 1][v + half * 8][4 + w * 16 + col] = (_Float16)(ov * ftanh(cn));
    }
    __syncthreads();                              // h_{t+1}/x_{t+1} complete for all 4 waves
  }

  // out[b] = h_n . W_fc + b_fc  (h_n sits in abuf[TSTEPS & 1] == abuf[0])
  if (w == 0 && lane < 16) {
    float acc = bfc[0];
    for (int j = 0; j < 64; ++j)
      acc += (float)abuf[0][lane][4 + j] * Wfc[j];
    out[b0 + lane] = acc;
  }
}

extern "C" void kernel_launch(void* const* d_in, const int* in_sizes, int n_in,
                              void* d_out, int out_size, void* d_ws, size_t ws_size,
                              hipStream_t stream) {
  (void)in_sizes; (void)n_in; (void)d_ws; (void)ws_size; (void)out_size;
  const float* x   = (const float*)d_in[0];
  const float* Wih = (const float*)d_in[1];
  const float* Whh = (const float*)d_in[2];
  const float* bih = (const float*)d_in[3];
  const float* bhh = (const float*)d_in[4];
  const float* Wfc = (const float*)d_in[5];
  const float* bfc = (const float*)d_in[6];
  float* out = (float*)d_out;

  const int B = 8192;                 // x: [8192, 512, 4]
  dim3 grid(B / 16);                  // 512 blocks, one 16-batch tile each
  dim3 block(WPB * 32);               // 4 waves split the 256 gate columns 4 ways
  firelstm_kernel<<<grid, block, 0, stream>>>(x, Wih, Whh, bih, bhh, Wfc, bfc, out);
}
